// EncoderLayer_90245852823738
// MI455X (gfx1250) — matrix operand order repaired and checked by reading, not compiled, not run on hardware
//
#include <hip/hip_runtime.h>
#include <hip/hip_bf16.h>

typedef __attribute__((ext_vector_type(16))) _Float16 v16h;
typedef __attribute__((ext_vector_type(8)))  float    v8f;

#define M_TOK   8192   // B*L
#define D_MODEL 512
#define D_FF    2048
#define N_HEADS 8
#define E_HEAD  64
#define SEQ_L   2048
#define BATCH   4

__device__ __forceinline__ v8f wmma_f16(v16h a, v16h b, v8f c) {
  return __builtin_amdgcn_wmma_f32_16x16x32_f16(false, a, false, b, (short)0, c, false, false);
}

// Async copy 32 bytes (2x b128) global -> LDS. lds is the byte offset of the
// destination inside the wave's LDS window (low 32 bits of a generic pointer
// to __shared__); the instruction offset applies to both sides.
__device__ __forceinline__ void async_b128x2(uint32_t lds, const void* g) {
  asm volatile("global_load_async_to_lds_b128 %0, %1, off\n\t"
               "global_load_async_to_lds_b128 %0, %1, off offset:16"
               :: "v"(lds), "v"(g) : "memory");
}
__device__ __forceinline__ void wait_async0() {
  asm volatile("s_wait_asynccnt 0" ::: "memory");
}
__device__ __forceinline__ uint32_t lds_off(const void* p) {
  return (uint32_t)(uintptr_t)p;
}

// A fragment (16x32 f16) from row-major [*, LD] f16 buffer.
template<int LD>
__device__ __forceinline__ v16h frag_a(const _Float16* b, int row0, int k0, int lane) {
  int r  = row0 + (lane & 15);
  int kb = k0 + ((lane >> 4) << 4);
  union { uint4 u[2]; v16h v; } x;
  x.u[0] = *reinterpret_cast<const uint4*>(b + r * LD + kb);
  x.u[1] = *reinterpret_cast<const uint4*>(b + r * LD + kb + 8);
  return x.v;
}

// B fragment (32x16 f16): buffer holds [col][k] rows with leading dim LD.
template<int LD>
__device__ __forceinline__ v16h frag_b(const _Float16* b, int col0, int k0, int lane) {
  int c  = col0 + (lane & 15);
  int kb = k0 + ((lane >> 4) << 4);
  const uint4* p = reinterpret_cast<const uint4*>(b + c * LD + kb);
  union { uint4 u[2]; v16h v; } x;
  x.u[0] = p[0]; x.u[1] = p[1];
  return x.v;
}

// ---------------------------------------------------------------- converts
__global__ __launch_bounds__(256) void cvt_f32_f16(const float* __restrict__ in,
                                                   _Float16* __restrict__ out, int n) {
  int i = blockIdx.x * 256 + threadIdx.x;
  if (i < n) out[i] = (_Float16)in[i];
}

// in: [K][N] f32 -> out: [N][K] f16 (transposed weights)
__global__ __launch_bounds__(256) void cvt_f32_f16_T(const float* __restrict__ in,
                                                     _Float16* __restrict__ out,
                                                     int K, int N) {
  int i = blockIdx.x * 256 + threadIdx.x;
  if (i < K * N) {
    int k = i / N, n = i - k * N;
    out[(size_t)n * K + k] = (_Float16)in[i];
  }
}

// ---------------------------------------------------------------- GEMM
// C[M,N] = act(A[M,K] @ B[K,N] + bias), with B supplied TRANSPOSED: BT[N][K].
// 128x128 block tile, BK=32, double-buffered async LDS staging.
// OUT_MODE: 0 = f32, 1 = f16, 2 = f16 transposed-per-batch [B][N][L] (for V).
template<int OUT_MODE, int GELU>
__global__ __launch_bounds__(256) void gemm_kernel(
    const _Float16* __restrict__ A, const _Float16* __restrict__ BT,
    const float* __restrict__ bias,
    float* __restrict__ Cf, _Float16* __restrict__ Ch,
    int M, int N, int K) {
  __shared__ __attribute__((aligned(16))) _Float16 Asm[2][128 * 32];
  __shared__ __attribute__((aligned(16))) _Float16 Bsm[2][128 * 32]; // [n][k]

  const int tid  = threadIdx.x;
  const int lane = tid & 31;
  const int wave = tid >> 5;
  const int wm   = wave & 3;   // 4 row groups of 32
  const int wn   = wave >> 2;  // 2 col groups of 64
  const int m0   = blockIdx.y * 128;
  const int n0   = blockIdx.x * 128;

  const int srow = tid >> 1;          // 0..127
  const int skh  = (tid & 1) * 16;    // 0 or 16 halves

  auto stage = [&](int kt, int bi) {
    async_b128x2(lds_off(&Asm[bi][srow * 32 + skh]),
                 A + (size_t)(m0 + srow) * K + kt + skh);
    async_b128x2(lds_off(&Bsm[bi][srow * 32 + skh]),
                 BT + (size_t)(n0 + srow) * K + kt + skh);
  };

  const v8f vzero = {};
  v8f acc[2][4];
#pragma unroll
  for (int i = 0; i < 2; ++i)
#pragma unroll
    for (int j = 0; j < 4; ++j) acc[i][j] = vzero;

  stage(0, 0);
  wait_async0();
  __syncthreads();

  int buf = 0;
  for (int kt = 0; kt < K; kt += 32) {
    if (kt + 32 < K) stage(kt + 32, buf ^ 1);

    v16h af[2], bf[4];
#pragma unroll
    for (int mt = 0; mt < 2; ++mt) af[mt] = frag_a<32>(Asm[buf], wm * 32 + mt * 16, 0, lane);
#pragma unroll
    for (int nt = 0; nt < 4; ++nt) bf[nt] = frag_b<32>(Bsm[buf], wn * 64 + nt * 16, 0, lane);
#pragma unroll
    for (int mt = 0; mt < 2; ++mt)
#pragma unroll
      for (int nt = 0; nt < 4; ++nt) acc[mt][nt] = wmma_f16(af[mt], bf[nt], acc[mt][nt]);

    wait_async0();
    __syncthreads();
    buf ^= 1;
  }

  // epilogue
#pragma unroll
  for (int mt = 0; mt < 2; ++mt)
#pragma unroll
    for (int nt = 0; nt < 4; ++nt) {
      int colb = n0 + wn * 64 + nt * 16 + (lane & 15);
      float bv = bias ? bias[colb] : 0.0f;
#pragma unroll
      for (int r = 0; r < 8; ++r) {
        int gr = m0 + wm * 32 + mt * 16 + ((lane >> 4) << 3) + r;
        float v = acc[mt][nt][r] + bv;
        if (GELU) v = 0.5f * v * (1.0f + erff(v * 0.70710678118654752f));
        if (OUT_MODE == 0) {
          Cf[(size_t)gr * N + colb] = v;
        } else if (OUT_MODE == 1) {
          Ch[(size_t)gr * N + colb] = (_Float16)v;
        } else { // V projection: write [B][N][L] so attention can async-copy V^T tiles
          int b = gr >> 11;            // / SEQ_L
          int l = gr & (SEQ_L - 1);
          Ch[((size_t)b * D_MODEL + colb) * SEQ_L + l] = (_Float16)v;
        }
      }
    }
}

// ---------------------------------------------------------------- flash attention
// grid.x = L/128 (query blocks), grid.y = B*H. Each wave owns 16 query rows.
// Q,K: [B*L, D] f16 row-major.  Vt: [B][D][L] f16 (pre-transposed).
__global__ __launch_bounds__(256) void attn_kernel(
    const _Float16* __restrict__ Q, const _Float16* __restrict__ Kh,
    const _Float16* __restrict__ Vt, _Float16* __restrict__ O) {
  __shared__ __attribute__((aligned(16))) _Float16 Qs[128 * 64];
  __shared__ __attribute__((aligned(16))) _Float16 Ks[2][64 * 64];   // [key][e]
  __shared__ __attribute__((aligned(16))) _Float16 Vs[2][64 * 64];   // [e][key]
  __shared__ __attribute__((aligned(16))) _Float16 Ps[128 * 64];     // [qrow][key]

  const int tid  = threadIdx.x;
  const int lane = tid & 31;
  const int wave = tid >> 5;
  const int ln   = lane & 15;
  const int lh   = lane >> 4;

  const int bh = blockIdx.y;
  const int b  = bh / N_HEADS;
  const int hd = bh % N_HEADS;
  const int q0 = blockIdx.x * 128;
  const size_t headOff = (size_t)b * SEQ_L * D_MODEL + hd * E_HEAD;
  const _Float16* Vhead = Vt + ((size_t)b * D_MODEL + hd * E_HEAD) * SEQ_L;

  // stage Q tile (async): 128 x 64
  {
    int row = tid >> 1, off = (tid & 1) * 32;
    const _Float16* g = Q + headOff + (size_t)(q0 + row) * D_MODEL + off;
    uint32_t l = lds_off(&Qs[row * 64 + off]);
    async_b128x2(l, g);
    async_b128x2(l + 32, g + 16);
  }

  const int skey = tid >> 2;          // 0..63
  const int seo  = (tid & 3) * 16;    // 0,16,32,48
  auto stageKV = [&](int sc, int bi) {
    // K chunk: direct rows [key][e]
    async_b128x2(lds_off(&Ks[bi][skey * 64 + seo]),
                 Kh + headOff + (size_t)(sc * 64 + skey) * D_MODEL + seo);
    // V chunk: direct rows of pre-transposed V -> [e][key]
    async_b128x2(lds_off(&Vs[bi][skey * 64 + seo]),
                 Vhead + (size_t)skey * SEQ_L + sc * 64 + seo);
  };

  stageKV(0, 0);
  wait_async0();
  __syncthreads();

  v16h aq[2];
#pragma unroll
  for (int kk = 0; kk < 2; ++kk) aq[kk] = frag_a<64>(Qs, wave * 16, kk * 32, lane);

  const v8f vzero = {};
  v8f ov[4];
#pragma unroll
  for (int i = 0; i < 4; ++i) ov[i] = vzero;
  float mrow[8], lrow[8];
#pragma unroll
  for (int r = 0; r < 8; ++r) { mrow[r] = -1.0e30f; lrow[r] = 0.0f; }

  const float scale = 0.125f; // 1/sqrt(64)
  const int nChunks = SEQ_L / 64;

  int buf = 0;
  for (int sc = 0; sc < nChunks; ++sc) {
    if (sc + 1 < nChunks) stageKV(sc + 1, buf ^ 1);

    // S = Q @ K^T  (16 rows x 64 keys per wave)
    v8f s[4];
#pragma unroll
    for (int nt = 0; nt < 4; ++nt) s[nt] = vzero;
#pragma unroll
    for (int kk = 0; kk < 2; ++kk)
#pragma unroll
      for (int nt = 0; nt < 4; ++nt)
        s[nt] = wmma_f16(aq[kk], frag_b<64>(Ks[buf], nt * 16, kk * 32, lane), s[nt]);

    // online softmax: lanes 0-15 hold row r, lanes 16-31 row r+8;
    // xor masks <=8 reduce within each 16-lane half.
#pragma unroll
    for (int r = 0; r < 8; ++r) {
      float m = -1.0e30f;
#pragma unroll
      for (int nt = 0; nt < 4; ++nt) {
        s[nt][r] *= scale;
        m = fmaxf(m, s[nt][r]);
      }
      m = fmaxf(m, __shfl_xor(m, 1));
      m = fmaxf(m, __shfl_xor(m, 2));
      m = fmaxf(m, __shfl_xor(m, 4));
      m = fmaxf(m, __shfl_xor(m, 8));
      float mnew = fmaxf(mrow[r], m);
      float corr = __expf(mrow[r] - mnew);
      mrow[r] = mnew;
      float rs = 0.0f;
#pragma unroll
      for (int nt = 0; nt < 4; ++nt) {
        float p = __expf(s[nt][r] - mnew);
        s[nt][r] = p;
        rs += p;
      }
      rs += __shfl_xor(rs, 1);
      rs += __shfl_xor(rs, 2);
      rs += __shfl_xor(rs, 4);
      rs += __shfl_xor(rs, 8);
      lrow[r] = lrow[r] * corr + rs;
#pragma unroll
      for (int nt = 0; nt < 4; ++nt) ov[nt][r] *= corr;
    }

    // write P (C-layout) to LDS, re-read as A fragments
#pragma unroll
    for (int nt = 0; nt < 4; ++nt)
#pragma unroll
      for (int r = 0; r < 8; ++r)
        Ps[(wave * 16 + (lh << 3) + r) * 64 + nt * 16 + ln] = (_Float16)s[nt][r];
    __syncthreads();

    v16h ap[2];
#pragma unroll
    for (int kk = 0; kk < 2; ++kk) ap[kk] = frag_a<64>(Ps, wave * 16, kk * 32, lane);
#pragma unroll
    for (int kk = 0; kk < 2; ++kk)
#pragma unroll
      for (int nt = 0; nt < 4; ++nt)
        ov[nt] = wmma_f16(ap[kk], frag_b<64>(Vs[buf], nt * 16, kk * 32, lane), ov[nt]);

    wait_async0();
    __syncthreads();
    buf ^= 1;
  }

  // write O / l
#pragma unroll
  for (int nt = 0; nt < 4; ++nt)
#pragma unroll
    for (int r = 0; r < 8; ++r) {
      int qrow = q0 + wave * 16 + (lh << 3) + r;
      float o  = ov[nt][r] / lrow[r];
      O[headOff + (size_t)qrow * D_MODEL + nt * 16 + ln] = (_Float16)o;
    }
}

// ---------------------------------------------------------------- residual + layernorm
__global__ __launch_bounds__(256) void add_ln_kernel(
    const float* __restrict__ X, const float* __restrict__ Yd,
    const float* __restrict__ g, const float* __restrict__ be,
    float* __restrict__ Of, _Float16* __restrict__ Oh) {
  const int wave = threadIdx.x >> 5, lane = threadIdx.x & 31;
  const size_t row = (size_t)blockIdx.x * 8 + wave;
  const float* x = X + row * D_MODEL;
  const float* y = Yd + row * D_MODEL;

  float v[16];
  float s = 0.0f;
#pragma unroll
  for (int i = 0; i < 16; ++i) {
    int e = i * 32 + lane;
    v[i] = x[e] + y[e];
    s += v[i];
  }
  s += __shfl_xor(s, 1);  s += __shfl_xor(s, 2);
  s += __shfl_xor(s, 4);  s += __shfl_xor(s, 8);  s += __shfl_xor(s, 16);
  float mu = s * (1.0f / (float)D_MODEL);
  float q = 0.0f;
#pragma unroll
  for (int i = 0; i < 16; ++i) { float d = v[i] - mu; q += d * d; }
  q += __shfl_xor(q, 1);  q += __shfl_xor(q, 2);
  q += __shfl_xor(q, 4);  q += __shfl_xor(q, 8);  q += __shfl_xor(q, 16);
  float inv = rsqrtf(q * (1.0f / (float)D_MODEL) + 1e-5f);
#pragma unroll
  for (int i = 0; i < 16; ++i) {
    int e = i * 32 + lane;
    float o = (v[i] - mu) * inv * g[e] + be[e];
    Of[row * D_MODEL + e] = o;
    if (Oh) Oh[row * D_MODEL + e] = (_Float16)o;
  }
}

// ---------------------------------------------------------------- launch
extern "C" void kernel_launch(void* const* d_in, const int* in_sizes, int n_in,
                              void* d_out, int out_size, void* d_ws, size_t ws_size,
                              hipStream_t stream) {
  const float* x  = (const float*)d_in[0];
  const float* Wq = (const float*)d_in[1];  const float* bq = (const float*)d_in[2];
  const float* Wk = (const float*)d_in[3];  const float* bk = (const float*)d_in[4];
  const float* Wv = (const float*)d_in[5];  const float* bv = (const float*)d_in[6];
  const float* Wo = (const float*)d_in[7];  const float* bo = (const float*)d_in[8];
  const float* W1 = (const float*)d_in[9];  const float* b1 = (const float*)d_in[10];
  const float* W2 = (const float*)d_in[11]; const float* b2 = (const float*)d_in[12];
  const float* g1 = (const float*)d_in[13]; const float* be1 = (const float*)d_in[14];
  const float* g2 = (const float*)d_in[15]; const float* be2 = (const float*)d_in[16];

  const size_t M = M_TOK, D = D_MODEL, F = D_FF;
  char* w = (char*)d_ws;
  size_t o = 0;
  auto take = [&](size_t bytes) { size_t r = o; o += (bytes + 255) & ~(size_t)255; return r; };

  size_t xh_o = take(M * D * 2);
  size_t wq_o = take(D * D * 2), wk_o = take(D * D * 2);
  size_t wv_o = take(D * D * 2), wo_o = take(D * D * 2);
  size_t w1_o = take(D * F * 2), w2_o = take(F * D * 2);
  size_t big_o = take(M * F * 2);   // overlay: q|k|vt|av, later reused for h
  size_t f1_o  = take(M * D * 4);   // attn_out f32, later reused for y f32
  size_t x1f_o = take(M * D * 4);
  size_t x1h_o = take(M * D * 2);

  _Float16* xh  = (_Float16*)(w + xh_o);
  _Float16* wqT = (_Float16*)(w + wq_o);
  _Float16* wkT = (_Float16*)(w + wk_o);
  _Float16* wvT = (_Float16*)(w + wv_o);
  _Float16* woT = (_Float16*)(w + wo_o);
  _Float16* w1T = (_Float16*)(w + w1_o);
  _Float16* w2T = (_Float16*)(w + w2_o);
  _Float16* qh  = (_Float16*)(w + big_o);
  _Float16* kh  = (_Float16*)(w + big_o + M * D * 2);
  _Float16* vt  = (_Float16*)(w + big_o + 2 * M * D * 2);
  _Float16* avh = (_Float16*)(w + big_o + 3 * M * D * 2);
  _Float16* hh  = (_Float16*)(w + big_o);       // reuse after attention path done
  float*    attnf = (float*)(w + f1_o);
  float*    yf    = (float*)(w + f1_o);         // reuse
  float*    x1f   = (float*)(w + x1f_o);
  _Float16* x1h   = (_Float16*)(w + x1h_o);

  // convert: x plain, weights transposed ([K][N] -> [N][K])
  cvt_f32_f16<<<(int)((M * D + 255) / 256), 256, 0, stream>>>(x, xh, (int)(M * D));
  cvt_f32_f16_T<<<(int)((D * D + 255) / 256), 256, 0, stream>>>(Wq, wqT, (int)D, (int)D);
  cvt_f32_f16_T<<<(int)((D * D + 255) / 256), 256, 0, stream>>>(Wk, wkT, (int)D, (int)D);
  cvt_f32_f16_T<<<(int)((D * D + 255) / 256), 256, 0, stream>>>(Wv, wvT, (int)D, (int)D);
  cvt_f32_f16_T<<<(int)((D * D + 255) / 256), 256, 0, stream>>>(Wo, woT, (int)D, (int)D);
  cvt_f32_f16_T<<<(int)((D * F + 255) / 256), 256, 0, stream>>>(W1, w1T, (int)D, (int)F);
  cvt_f32_f16_T<<<(int)((F * D + 255) / 256), 256, 0, stream>>>(W2, w2T, (int)F, (int)D);

  dim3 blk(256);
  // QKV projections (V written pre-transposed [B][D][L])
  gemm_kernel<1, 0><<<dim3(D / 128, M / 128), blk, 0, stream>>>(xh, wqT, bq, nullptr, qh, M, D, D);
  gemm_kernel<1, 0><<<dim3(D / 128, M / 128), blk, 0, stream>>>(xh, wkT, bk, nullptr, kh, M, D, D);
  gemm_kernel<2, 0><<<dim3(D / 128, M / 128), blk, 0, stream>>>(xh, wvT, bv, nullptr, vt, M, D, D);

  // flash attention
  attn_kernel<<<dim3(SEQ_L / 128, BATCH * N_HEADS), blk, 0, stream>>>(qh, kh, vt, avh);

  // output projection + residual + LN1
  gemm_kernel<0, 0><<<dim3(D / 128, M / 128), blk, 0, stream>>>(avh, woT, bo, attnf, nullptr, M, D, D);
  add_ln_kernel<<<(int)(M / 8), blk, 0, stream>>>(x, attnf, g1, be1, x1f, x1h);

  // FFN
  gemm_kernel<1, 1><<<dim3(F / 128, M / 128), blk, 0, stream>>>(x1h, w1T, b1, nullptr, hh, M, F, D);
  gemm_kernel<0, 0><<<dim3(D / 128, M / 128), blk, 0, stream>>>(hh, w2T, b2, yf, nullptr, M, D, F);
  add_ln_kernel<<<(int)(M / 8), blk, 0, stream>>>(x1f, yf, g2, be2, (float*)d_out, nullptr);
}
